// TokenEncoder_35648228557209
// MI455X (gfx1250) — matrix operand order repaired and checked by reading, not compiled, not run on hardware
//
#include <hip/hip_runtime.h>

// ---------------- problem constants ----------------
#define B_     4
#define L_     2048
#define DIN    256
#define E_     16
#define M_     768
#define NT     (B_ * L_)          // 8192 tokens
#define LP1    (L_ + 1)           // 2049
#define LISTN  (NT + E_ * 16)     // padded token list capacity = 8448
#define MAXTILES 528              // 8192/16 + 16

typedef __attribute__((ext_vector_type(16))) __bf16 v16bf;
typedef __attribute__((ext_vector_type(8)))  float  v8f;

union Frag16 { v16bf v; unsigned int u[8]; };

__device__ __forceinline__ unsigned short f2bf(float f) {
  union { float f; unsigned int u; } c; c.f = f;
  unsigned int u = c.u;
  u += 0x7FFFu + ((u >> 16) & 1u);         // round-to-nearest-even
  return (unsigned short)(u >> 16);
}

// ---------------- pass 0: W (E,K,M) f32 -> Wt (E,M,K) bf16 ----------------
__global__ void convert_w_kernel(const float* __restrict__ W,
                                 unsigned short* __restrict__ Wt) {
  int tid = blockIdx.x * blockDim.x + threadIdx.x;
  if (tid >= E_ * DIN * M_) return;
  int e   = tid / (DIN * M_);
  int rem = tid - e * (DIN * M_);
  int k   = rem / M_;
  int m   = rem - k * M_;
  Wt[((size_t)(e * M_ + m)) * DIN + k] = f2bf(W[tid]);
}

// ---------------- pass 1a: init workspace metadata ----------------
__global__ void init_meta_kernel(unsigned int* __restrict__ counts,
                                 unsigned int* __restrict__ counts2,
                                 int* __restrict__ tileExpert,
                                 int* __restrict__ tokenList) {
  int t = blockIdx.x * blockDim.x + threadIdx.x;
  if (t < LISTN)    tokenList[t]  = -1;
  if (t < MAXTILES) tileExpert[t] = -1;
  if (t < E_)       { counts[t] = 0u; counts2[t] = 0u; }
}

// ---------------- pass 1b: count tokens per expert ----------------
__global__ void count_kernel(const int* __restrict__ sid,
                             unsigned int* __restrict__ counts) {
  int t = blockIdx.x * blockDim.x + threadIdx.x;
  if (t < NT) atomicAdd(&counts[sid[t]], 1u);
}

// ---------------- pass 1c: prefix + tile plan (E=16, trivial) ----------------
__global__ void prefix_kernel(const unsigned int* __restrict__ counts,
                              unsigned int* __restrict__ offsets,
                              int* __restrict__ tileExpert,
                              int* __restrict__ tileRowStart) {
  if (blockIdx.x != 0 || threadIdx.x != 0) return;
  unsigned int off = 0; int tile = 0;
  for (int e = 0; e < E_; ++e) {
    offsets[e] = off;
    unsigned int c = counts[e];
    unsigned int tiles = (c + 15u) >> 4;
    for (unsigned int tt = 0; tt < tiles; ++tt) {
      tileExpert[tile]   = e;
      tileRowStart[tile] = (int)(off + tt * 16u);
      ++tile;
    }
    off += tiles * 16u;
  }
}

// ---------------- pass 1d: scatter token ids into expert buckets ----------------
__global__ void scatter_kernel(const int* __restrict__ sid,
                               const unsigned int* __restrict__ offsets,
                               unsigned int* __restrict__ counts2,
                               int* __restrict__ tokenList) {
  int t = blockIdx.x * blockDim.x + threadIdx.x;
  if (t >= NT) return;
  int e = sid[t];
  unsigned int p = atomicAdd(&counts2[e], 1u);
  tokenList[offsets[e] + p] = t;
}

// ---------------- pass 2: grouped WMMA GEMM + fused epilogue ----------------
__global__ __launch_bounds__(256)
void gemm_kernel(const int* __restrict__ tokenList,
                 const int* __restrict__ tileExpert,
                 const int* __restrict__ tileRowStart,
                 const float* __restrict__ emb,
                 const unsigned short* __restrict__ Wt,
                 const float* __restrict__ bias,
                 const float* __restrict__ pos_tab,
                 const float* __restrict__ id_tab,
                 const float* __restrict__ mod_tab,
                 const float* __restrict__ role_tab,
                 const int* __restrict__ pos,
                 const int* __restrict__ sid,
                 const int* __restrict__ mod,
                 const int* __restrict__ role,
                 const unsigned char* __restrict__ mask,
                 float* __restrict__ out) {
  const int tile = blockIdx.x;
  const int e = tileExpert[tile];
  if (e < 0) return;                               // block-uniform: EXEC stays all-1s
  const int rowStart = tileRowStart[tile];

  __shared__ alignas(16) unsigned short lA[16][DIN + 8];  // 16x256 bf16 A panel, padded
  __shared__ int sTok[16];

  const int t = threadIdx.x;                       // 0..255
  // ---- stage A panel: gather 16 token rows, f32 -> bf16 ----
  for (int i = 0; i < 16; ++i) {
    int tok = tokenList[rowStart + i];             // scalar per iteration
    float v = (tok >= 0) ? emb[(size_t)tok * DIN + t] : 0.0f;
    lA[i][t] = f2bf(v);
  }
  if (t < 16) sTok[t] = tokenList[rowStart + t];
  __syncthreads();

  const int lane   = t & 31;
  const int waveId = t >> 5;                       // 0..7, owns 96 columns
  const int half   = lane >> 4;                    // 0 | 1
  const int n      = lane & 15;

  v8f acc[6];
  for (int ct = 0; ct < 6; ++ct)
    for (int r = 0; r < 8; ++r) acc[ct][r] = 0.0f;

  const unsigned int* warow =
      reinterpret_cast<const unsigned int*>(&lA[n][0]);          // A row for this lane
  const int colBase = waveId * 96;

  // ---- K loop: 8 steps of 32 ----
  for (int ks = 0; ks < 8; ++ks) {
    // A fragment (16x32 bf16), ISA 7.12.2 layout
    Frag16 a;
    const int abase = ks * 16 + half * 4;          // uint32 index within row
    #pragma unroll
    for (int j = 0; j < 4; ++j) a.u[j]     = warow[abase + j];
    #pragma unroll
    for (int j = 0; j < 4; ++j) a.u[4 + j] = warow[abase + 8 + j];

    #pragma unroll
    for (int ct = 0; ct < 6; ++ct) {
      const int col = colBase + ct * 16 + n;
      const unsigned int* wp =
          reinterpret_cast<const unsigned int*>(Wt) +
          (size_t)(e * M_ + col) * (DIN / 2);
      const int bbase = ks * 16 + half * 4;        // K contiguous in Wt
      Frag16 b;
      #pragma unroll
      for (int j = 0; j < 8; ++j) b.u[j] = wp[bbase + (j & 3) + ((j >> 2) << 3)];
      acc[ct] = __builtin_amdgcn_wmma_f32_16x16x32_bf16(
          false, a.v, false, b.v, (short)0, acc[ct], false, false);
    }
  }

  // ---- fused epilogue: bias + mask + 4 table adds, scatter to tokens[:,1:,:] ----
  #pragma unroll
  for (int ct = 0; ct < 6; ++ct) {
    const int col = colBase + ct * 16 + n;
    const float bi = bias[e * M_ + col];
    #pragma unroll
    for (int r = 0; r < 8; ++r) {
      const int Mrow = r + half * 8;
      const int tok = sTok[Mrow];
      if (tok < 0) continue;
      float val = mask[tok] ? (acc[ct][r] + bi) : 0.0f;
      val += pos_tab[(size_t)pos[tok]  * M_ + col];
      val += id_tab [(size_t)sid[tok]  * M_ + col];
      val += mod_tab[(size_t)mod[tok]  * M_ + col];
      val += role_tab[(size_t)role[tok] * M_ + col];
      const int bidx = tok >> 11;                  // /L_
      const int lidx = tok & (L_ - 1);
      out[((size_t)(bidx * LP1 + lidx + 1)) * M_ + col] = val;
    }
  }
}

// ---------------- pass 3: CLS rows + attn_keep ----------------
__global__ void cls_attn_kernel(const float* __restrict__ cls_content,
                                const float* __restrict__ pos_tab,
                                const float* __restrict__ id_tab,
                                const unsigned char* __restrict__ mask,
                                float* __restrict__ out_tokens,
                                float* __restrict__ out_attn) {
  int idx = blockIdx.x * blockDim.x + threadIdx.x;
  if (idx < B_ * M_) {
    int b = idx / M_, c = idx - b * M_;
    out_tokens[(size_t)(b * LP1) * M_ + c] =
        cls_content[c] + pos_tab[c] + id_tab[(size_t)E_ * M_ + c];
  }
  if (idx < B_ * LP1) {
    int b = idx / LP1, p = idx - b * LP1;
    float keep = (p == 0) ? 1.0f : (mask[b * L_ + (p - 1)] ? 1.0f : 0.0f);
    out_attn[b * LP1 + p] = keep;
  }
}

// ---------------- host-side launch ----------------
extern "C" void kernel_launch(void* const* d_in, const int* in_sizes, int n_in,
                              void* d_out, int out_size, void* d_ws, size_t ws_size,
                              hipStream_t stream) {
  const float* emb      = (const float*)d_in[0];
  const float* W        = (const float*)d_in[1];
  const float* bias     = (const float*)d_in[2];
  const float* cls      = (const float*)d_in[3];
  const float* pos_tab  = (const float*)d_in[4];
  const float* id_tab   = (const float*)d_in[5];
  const float* mod_tab  = (const float*)d_in[6];
  const float* role_tab = (const float*)d_in[7];
  const int*   pos      = (const int*)d_in[8];
  const int*   sid      = (const int*)d_in[9];
  const int*   mod      = (const int*)d_in[10];
  const int*   role     = (const int*)d_in[11];
  const unsigned char* mask = (const unsigned char*)d_in[12];

  float* out_tokens = (float*)d_out;
  float* out_attn   = (float*)d_out + (size_t)B_ * LP1 * M_;

  char* ws = (char*)d_ws;
  const size_t WT_BYTES = (size_t)E_ * M_ * DIN * 2;       // 6,291,456
  unsigned short* Wt      = (unsigned short*)ws;
  unsigned int* counts    = (unsigned int*)(ws + WT_BYTES);
  unsigned int* counts2   = (unsigned int*)(ws + WT_BYTES + 64);
  unsigned int* offsets   = (unsigned int*)(ws + WT_BYTES + 128);
  int* tileExpert         = (int*)(ws + WT_BYTES + 192);
  int* tileRowStart       = (int*)(ws + WT_BYTES + 192 + MAXTILES * 4);
  int* tokenList          = (int*)(ws + WT_BYTES + 192 + 2 * MAXTILES * 4);

  // pass 0: weight convert/transpose (f32 -> bf16, K-major)
  convert_w_kernel<<<(E_ * DIN * M_ + 255) / 256, 256, 0, stream>>>(W, Wt);
  // pass 1: expert binning
  init_meta_kernel<<<(LISTN + 255) / 256, 256, 0, stream>>>(counts, counts2,
                                                            tileExpert, tokenList);
  count_kernel<<<(NT + 255) / 256, 256, 0, stream>>>(sid, counts);
  prefix_kernel<<<1, 32, 0, stream>>>(counts, offsets, tileExpert, tileRowStart);
  scatter_kernel<<<(NT + 255) / 256, 256, 0, stream>>>(sid, offsets, counts2,
                                                       tokenList);
  // pass 2: grouped WMMA GEMM + fused epilogue
  gemm_kernel<<<MAXTILES, 256, 0, stream>>>(tokenList, tileExpert, tileRowStart,
                                            emb, Wt, bias, pos_tab, id_tab,
                                            mod_tab, role_tab, pos, sid, mod,
                                            role, mask, out_tokens);
  // pass 3: CLS + attn mask
  int n3 = (B_ * LP1 > B_ * M_) ? B_ * LP1 : B_ * M_;
  cls_attn_kernel<<<(n3 + 255) / 256, 256, 0, stream>>>(cls, pos_tab, id_tab,
                                                        mask, out_tokens, out_attn);
}
